// CLIPVisionTowerContextMoe_14362370638250
// MI455X (gfx1250) — compile-verified, matrix operand order checked by reading, not run on hardware
//
#include <hip/hip_runtime.h>
#include <hip/hip_bf16.h>

// ---------------------------------------------------------------------------
// CLIPVisionTowerContextMoe for MI455X (gfx1250, wave32, WMMA).
// Compute-bound (~750 GFLOP vs ~400MB): all large GEMMs go through
// v_wmma_f32_16x16x32_bf16 with f32 accumulation. bf16 conversion done once.
// GEMM: double-buffered LDS; next chunk's global loads are issued BEFORE the
// current chunk's WMMAs and stored to LDS AFTER them, so s_wait_loadcnt lands
// behind the matrix math. amdgpu_waves_per_eu(4,4) (= exactly one 8-wave
// workgroup per CU in CU mode) raises the VGPR cap so the staging registers
// stay resident -- no scratch spills.
// ---------------------------------------------------------------------------

typedef __bf16 bf16_t;
typedef __attribute__((ext_vector_type(16))) __bf16 v16bf;
typedef __attribute__((ext_vector_type(8)))  float  v8f;

#define B_    16
#define NTOK  577
#define D_    1024
#define C_    768
#define E_    4
#define F_    4096
#define H_    4
#define M_    (B_ * NTOK)   // 9232
#define KD_   (D_ + C_)     // 1792

__device__ __forceinline__ float qgelu_f(float x) {
  // x * sigmoid(1.702 x)
  return x / (1.0f + __expf(-1.702f * x));
}

// ---------------------------------------------------------------------------
// LDS fragment loader for 16-bit WMMA operands (A 16x32 / B 32x16 layouts):
// lanes 0-15 hold K = {0..7, 16..23}, lanes 16-31 hold K = {8..15, 24..31},
// per CDNA5 ISA "16-bit A-Matrix 16x32" table.
// ---------------------------------------------------------------------------
__device__ __forceinline__ v16bf load_frag_lds(const bf16_t* base, int row0,
                                               int stride, int lane) {
  int l  = lane & 15;
  int hi = lane >> 4;
  const bf16_t* p = base + (row0 + l) * stride + hi * 8;
  union { int4 i4[2]; v16bf v; } u;
  u.i4[0] = *(const int4*)(p);
  u.i4[1] = *(const int4*)(p + 16);
  return u.v;
}

enum { EPI_QGELU_BF16 = 0, EPI_ADDRES_F32 = 1, EPI_F32 = 2, EPI_MOE = 3 };

// ---------------------------------------------------------------------------
// Tiled bf16 GEMM: out[M,N] = A[M,K] * W[N,K]^T (+ epilogue).
// Block tile 128x128, K-chunk 32, double-buffered LDS, 1 barrier/chunk.
// 256 threads = 8 waves, wave grid 4x2, each wave owns 32x64 => 8 WMMAs.
// A-tile rows past M are CLAMPED (not zero-filled): tile row r only feeds
// output row r, which the epilogue masks, so no divergent guard is needed.
// ---------------------------------------------------------------------------
template<int EPI, bool FIRST>
__global__
__attribute__((amdgpu_flat_work_group_size(256, 256), amdgpu_waves_per_eu(4, 4)))
void gemm_wmma_bf16(const bf16_t* __restrict__ A, const bf16_t* __restrict__ W,
                    const float* __restrict__ bias, float* __restrict__ outF,
                    bf16_t* __restrict__ outB, const float* __restrict__ res,
                    const float* __restrict__ gate, int eidx,
                    int M, int N, int K)
{
  constexpr int LS = 40;  // 32 + 8 bf16 pad (80B rows, 16B aligned)
  __shared__ __align__(16) bf16_t Asm[2][128 * LS];
  __shared__ __align__(16) bf16_t Bsm[2][128 * LS];

  const int tid  = threadIdx.x;
  const int lane = tid & 31;
  const int wave = tid >> 5;
  const int wm   = wave >> 1;     // 0..3 -> 32-row slab
  const int wn   = wave & 1;      // 0..1 -> 64-col slab
  const int m0   = blockIdx.y * 128;
  const int n0   = blockIdx.x * 128;

  // Per-thread staging coordinates: 2 16B chunks each for A and B.
  int rr[2], cc[2];
  const bf16_t* pa[2];
  const bf16_t* pw[2];
#pragma unroll
  for (int u = 0; u < 2; ++u) {
    int t = tid + u * 256;            // 0..511
    rr[u] = t >> 2;                   // tile row 0..127
    cc[u] = (t & 3) << 3;             // k-offset {0,8,16,24}
    int gr = m0 + rr[u];
    if (gr >= M) gr = M - 1;          // clamp: dead rows masked in epilogue
    pa[u] = A + (size_t)gr * K + cc[u];
    pw[u] = W + (size_t)(n0 + rr[u]) * K + cc[u];
  }

  v8f acc[2][4] = {};

  // Preload chunk 0 into buffer 0.
#pragma unroll
  for (int u = 0; u < 2; ++u) {
    *(int4*)(&Asm[0][rr[u] * LS + cc[u]]) = *(const int4*)(pa[u]);
    *(int4*)(&Bsm[0][rr[u] * LS + cc[u]]) = *(const int4*)(pw[u]);
  }
  __syncthreads();

  const int nsteps = K >> 5;
  for (int s = 0; s < nsteps; ++s) {
    const int p = s & 1;
    const bool more = (s + 1 < nsteps);

    // (1) Issue next chunk's global loads into registers (no wait yet).
    int4 av[2], bv[2];
    if (more) {
      int kc = (s + 1) << 5;
#pragma unroll
      for (int u = 0; u < 2; ++u) {
        av[u] = *(const int4*)(pa[u] + kc);
        bv[u] = *(const int4*)(pw[u] + kc);
      }
    }

    // (2) Fragment loads (DScnt only) + 8 WMMAs on the current buffer.
    v16bf af[2], bfr[4];
#pragma unroll
    for (int i = 0; i < 2; ++i)
      af[i] = load_frag_lds(Asm[p], wm * 32 + i * 16, LS, lane);
#pragma unroll
    for (int j = 0; j < 4; ++j)
      bfr[j] = load_frag_lds(Bsm[p], wn * 64 + j * 16, LS, lane);

#pragma unroll
    for (int i = 0; i < 2; ++i)
#pragma unroll
      for (int j = 0; j < 4; ++j)
        acc[i][j] = __builtin_amdgcn_wmma_f32_16x16x32_bf16(
            false, af[i], false, bfr[j], (short)0, acc[i][j], false, false);

    // (3) Store staged registers to the other buffer -- the s_wait_loadcnt
    // for the global loads lands here, AFTER the WMMAs were issued.
    if (more) {
#pragma unroll
      for (int u = 0; u < 2; ++u) {
        *(int4*)(&Asm[p ^ 1][rr[u] * LS + cc[u]]) = av[u];
        *(int4*)(&Bsm[p ^ 1][rr[u] * LS + cc[u]]) = bv[u];
      }
    }

    // (4) One barrier per chunk: covers "writes to buf p^1 done" and
    // "reads from buf p done" before p is overwritten next iteration.
    __syncthreads();
  }

  // Epilogue. C/D layout: lane&15 = N, vgpr r = M row (+8 for lanes 16-31).
  const int ln_ = lane & 15;
  const int hi  = lane >> 4;
#pragma unroll
  for (int i = 0; i < 2; ++i) {
#pragma unroll
    for (int j = 0; j < 4; ++j) {
      int gn = n0 + wn * 64 + j * 16 + ln_;
      float bv = bias ? bias[gn] : 0.0f;
#pragma unroll
      for (int r = 0; r < 8; ++r) {
        int gm = m0 + wm * 32 + i * 16 + hi * 8 + r;
        if (gm >= M) continue;
        float v = acc[i][j][r] + bv;
        size_t oidx = (size_t)gm * N + gn;
        if (EPI == EPI_QGELU_BF16) {
          outB[oidx] = (bf16_t)qgelu_f(v);
        } else if (EPI == EPI_ADDRES_F32) {
          outF[oidx] = v + res[oidx];
        } else if (EPI == EPI_F32) {
          outF[oidx] = v;
        } else { // EPI_MOE
          float g = gate[(gm / NTOK) * E_ + eidx];
          float contrib = g * v;
          if (FIRST) outF[oidx] = contrib;
          else       outF[oidx] += contrib;
        }
      }
    }
  }
}

// --------------------------- elementwise helpers ---------------------------

__global__ __launch_bounds__(256)
void cvt_bf16(const float* __restrict__ in, bf16_t* __restrict__ out, size_t n) {
  size_t stride = (size_t)gridDim.x * 256u;
  for (size_t i = (size_t)blockIdx.x * 256u + threadIdx.x; i < n; i += stride)
    out[i] = (bf16_t)in[i];
}

// Row LayerNorm; writes fp32 (stride Dd) and/or bf16 (stride obstride).
__global__ __launch_bounds__(256)
void ln_row(const float* __restrict__ x, const float* __restrict__ g,
            const float* __restrict__ b, float* __restrict__ outF,
            bf16_t* __restrict__ outB, int Dd, size_t obstride)
{
  __shared__ float red[256];
  int m = blockIdx.x, tid = threadIdx.x;
  const float* row = x + (size_t)m * Dd;

  float s = 0.0f;
  for (int d = tid; d < Dd; d += 256) s += row[d];
  red[tid] = s; __syncthreads();
  for (int o = 128; o > 0; o >>= 1) { if (tid < o) red[tid] += red[tid + o]; __syncthreads(); }
  float mean = red[0] / (float)Dd;
  __syncthreads();

  float v = 0.0f;
  for (int d = tid; d < Dd; d += 256) { float t = row[d] - mean; v += t * t; }
  red[tid] = v; __syncthreads();
  for (int o = 128; o > 0; o >>= 1) { if (tid < o) red[tid] += red[tid + o]; __syncthreads(); }
  float rstd = rsqrtf(red[0] / (float)Dd + 1e-5f);

  for (int d = tid; d < Dd; d += 256) {
    float y = (row[d] - mean) * rstd * g[d] + b[d];
    if (outF) outF[(size_t)m * Dd + d] = y;
    if (outB) outB[(size_t)m * obstride + d] = (bf16_t)y;
  }
}

// Broadcast txt features (bf16) into fused[:, Doff:Doff+C].
__global__ __launch_bounds__(256)
void bcast_txt(const float* __restrict__ txtf, bf16_t* __restrict__ fused,
               int Mrows, int ntok, int Cc, int FD, int Doff)
{
  size_t total  = (size_t)Mrows * Cc;
  size_t stride = (size_t)gridDim.x * 256u;
  for (size_t i = (size_t)blockIdx.x * 256u + threadIdx.x; i < total; i += stride) {
    int m = (int)(i / Cc), c = (int)(i % Cc);
    fused[(size_t)m * FD + Doff + c] = (bf16_t)txtf[(size_t)(m / ntok) * Cc + c];
  }
}

// Tiny fp32 linear: out[r,o] = act(in[r,:].w[o,:] + bias[o]) (+res). One
// thread per output element; only used for the negligible txt/attn pieces.
template<int ACT>
__global__ __launch_bounds__(256)
void small_linear(const float* __restrict__ in, const float* __restrict__ w,
                  const float* __restrict__ bias, const float* __restrict__ res,
                  float* __restrict__ out, int R, int O, int Kd)
{
  int idx = blockIdx.x * 256 + threadIdx.x;
  if (idx >= R * O) return;
  int r = idx / O, o = idx % O;
  const float* iv = in + (size_t)r * Kd;
  const float* wv = w  + (size_t)o * Kd;
  float s = bias ? bias[o] : 0.0f;
  for (int k = 0; k < Kd; ++k) s += iv[k] * wv[k];
  if (ACT == 1) s = qgelu_f(s);
  if (res) s += res[(size_t)r * O + o];
  out[(size_t)r * O + o] = s;
}

// Single-query cross-attention: one block per (b,h); softmax over N=577.
__global__ __launch_bounds__(256)
void attn_kernel(const float* __restrict__ q, const float* __restrict__ Kb,
                 const float* __restrict__ Vb, float* __restrict__ task,
                 int ntok, int Cc, int Hh)
{
  __shared__ float sc[608];
  __shared__ float red[256];
  int b = blockIdx.x / Hh, h = blockIdx.x % Hh;
  int hd = Cc / Hh;                       // 192
  int tid = threadIdx.x;
  float scale = rsqrtf((float)hd);
  const float* qv = q + (size_t)b * Cc + h * hd;

  for (int n = tid; n < ntok; n += 256) {
    const float* kv = Kb + ((size_t)(b * ntok + n)) * Cc + h * hd;
    float s = 0.0f;
    for (int d = 0; d < hd; ++d) s += qv[d] * kv[d];
    sc[n] = s * scale;
  }
  __syncthreads();

  float mx = -1e30f;
  for (int n = tid; n < ntok; n += 256) mx = fmaxf(mx, sc[n]);
  red[tid] = mx; __syncthreads();
  for (int o = 128; o > 0; o >>= 1) { if (tid < o) red[tid] = fmaxf(red[tid], red[tid + o]); __syncthreads(); }
  mx = red[0];
  __syncthreads();

  float sum = 0.0f;
  for (int n = tid; n < ntok; n += 256) { float e = __expf(sc[n] - mx); sc[n] = e; sum += e; }
  red[tid] = sum; __syncthreads();
  for (int o = 128; o > 0; o >>= 1) { if (tid < o) red[tid] += red[tid + o]; __syncthreads(); }
  float inv = 1.0f / red[0];
  __syncthreads();

  for (int d = tid; d < hd; d += 256) {
    float o = 0.0f;
    for (int n = 0; n < ntok; ++n)
      o += sc[n] * Vb[((size_t)(b * ntok + n)) * Cc + h * hd + d];
    task[(size_t)b * Cc + h * hd + d] = o * inv;
  }
}

__global__ void gate_softmax(const float* __restrict__ logits,
                             float* __restrict__ gp, int Bn, int Ee)
{
  int b = threadIdx.x;
  if (b >= Bn) return;
  float mx = -1e30f;
  for (int e = 0; e < Ee; ++e) mx = fmaxf(mx, logits[b * Ee + e]);
  float s = 0.0f;
  for (int e = 0; e < Ee; ++e) { float v = __expf(logits[b * Ee + e] - mx); gp[b * Ee + e] = v; s += v; }
  float inv = 1.0f / s;
  for (int e = 0; e < Ee; ++e) gp[b * Ee + e] *= inv;
}

// ---------------------------------------------------------------------------

extern "C" void kernel_launch(void* const* d_in, const int* in_sizes, int n_in,
                              void* d_out, int out_size, void* d_ws, size_t ws_size,
                              hipStream_t stream) {
  (void)in_sizes; (void)n_in; (void)out_size; (void)ws_size;
  const float* hid   = (const float*)d_in[0];
  const float* ctx   = (const float*)d_in[1];
  const float* ifc1w = (const float*)d_in[2];
  const float* ifc1b = (const float*)d_in[3];
  const float* ifc2w = (const float*)d_in[4];
  const float* ifc2b = (const float*)d_in[5];
  const float* ilng  = (const float*)d_in[6];
  const float* ilnb  = (const float*)d_in[7];
  const float* tfc1w = (const float*)d_in[8];
  const float* tfc1b = (const float*)d_in[9];
  const float* tfc2w = (const float*)d_in[10];
  const float* tfc2b = (const float*)d_in[11];
  const float* tlng  = (const float*)d_in[12];
  const float* tlnb  = (const float*)d_in[13];
  const float* qw    = (const float*)d_in[14];
  const float* qb    = (const float*)d_in[15];
  const float* kw    = (const float*)d_in[16];
  const float* kb    = (const float*)d_in[17];
  const float* vw    = (const float*)d_in[18];
  const float* vbp   = (const float*)d_in[19];
  const float* aow   = (const float*)d_in[20];
  const float* aob   = (const float*)d_in[21];
  const float* gw    = (const float*)d_in[22];
  const float* gb    = (const float*)d_in[23];
  const float* ew1   = (const float*)d_in[24];
  const float* eb1   = (const float*)d_in[25];
  const float* ew2   = (const float*)d_in[26];
  const float* eb2   = (const float*)d_in[27];

  float* out         = (float*)d_out;
  float* gate_logits = out + (size_t)M_ * D_;

  // ---- carve workspace ----
  char* ws = (char*)d_ws;
  size_t off = 0;
  auto carve = [&](size_t bytes) -> char* {
    size_t o = (off + 255) & ~(size_t)255;
    off = o + bytes;
    return ws + o;
  };
  bf16_t* hidb   = (bf16_t*)carve((size_t)M_ * D_ * 2);         // hidden bf16
  bf16_t* wfc1b  = (bf16_t*)carve((size_t)2 * D_ * D_ * 2);
  bf16_t* wfc2b  = (bf16_t*)carve((size_t)D_ * 2 * D_ * 2);
  bf16_t* wkb    = (bf16_t*)carve((size_t)C_ * KD_ * 2);
  bf16_t* wvb    = (bf16_t*)carve((size_t)C_ * KD_ * 2);
  bf16_t* we1b   = (bf16_t*)carve((size_t)E_ * F_ * D_ * 2);
  bf16_t* we2b   = (bf16_t*)carve((size_t)E_ * D_ * F_ * 2);
  bf16_t* hbig   = (bf16_t*)carve((size_t)M_ * F_ * 2);         // fc1-hidden & expert-hidden
  float*  preln  = (float*)carve((size_t)M_ * D_ * 4);          // pre-LN; reused as K buffer
  float*  vbuf   = (float*)carve((size_t)M_ * C_ * 4);
  bf16_t* fusedb = (bf16_t*)carve((size_t)M_ * KD_ * 2);
  float*  txth   = (float*)carve((size_t)16 * 1536 * 4);
  float*  txtpre = (float*)carve((size_t)16 * 768 * 4);
  float*  txtf   = (float*)carve((size_t)16 * 768 * 4);
  float*  qbuf   = (float*)carve((size_t)16 * 768 * 4);
  float*  taskp  = (float*)carve((size_t)16 * 768 * 4);
  float*  task   = (float*)carve((size_t)16 * 768 * 4);
  float*  gp     = (float*)carve((size_t)64 * 4);
  float*  kbuf   = preln;  // reuse (consumed by ln_row before K-proj runs)

  auto cvt = [&](const float* src, bf16_t* dst, size_t n) {
    int blocks = (int)((n + 255) / 256);
    if (blocks > 2048) blocks = 2048;
    cvt_bf16<<<blocks, 256, 0, stream>>>(src, dst, n);
  };

  // ---- fp32 -> bf16 conversions (one-time per call) ----
  cvt(hid,   hidb,  (size_t)M_ * D_);
  cvt(ifc1w, wfc1b, (size_t)2 * D_ * D_);
  cvt(ifc2w, wfc2b, (size_t)D_ * 2 * D_);
  cvt(kw,    wkb,   (size_t)C_ * KD_);
  cvt(vw,    wvb,   (size_t)C_ * KD_);
  cvt(ew1,   we1b,  (size_t)E_ * F_ * D_);
  cvt(ew2,   we2b,  (size_t)E_ * D_ * F_);

  const int mb = (M_ + 127) / 128;  // 73
  dim3 blk(256);

  // ---- image ResidualBlock ----
  gemm_wmma_bf16<EPI_QGELU_BF16, false><<<dim3(2 * D_ / 128, mb), blk, 0, stream>>>(
      hidb, wfc1b, ifc1b, nullptr, hbig, nullptr, nullptr, 0, M_, 2 * D_, D_);
  gemm_wmma_bf16<EPI_ADDRES_F32, false><<<dim3(D_ / 128, mb), blk, 0, stream>>>(
      hbig, wfc2b, ifc2b, preln, nullptr, hid, nullptr, 0, M_, D_, 2 * D_);
  ln_row<<<M_, 256, 0, stream>>>(preln, ilng, ilnb, nullptr, fusedb, D_, (size_t)KD_);

  // ---- text ResidualBlock (tiny) ----
  small_linear<1><<<(16 * 1536 + 255) / 256, 256, 0, stream>>>(
      ctx, tfc1w, tfc1b, nullptr, txth, 16, 1536, 768);
  small_linear<0><<<(16 * 768 + 255) / 256, 256, 0, stream>>>(
      txth, tfc2w, tfc2b, ctx, txtpre, 16, 768, 1536);
  ln_row<<<16, 256, 0, stream>>>(txtpre, tlng, tlnb, txtf, nullptr, C_, 0);
  bcast_txt<<<2048, 256, 0, stream>>>(txtf, fusedb, M_, NTOK, C_, KD_, D_);

  // ---- K / V projections over fused [M, 1792] ----
  gemm_wmma_bf16<EPI_F32, false><<<dim3(C_ / 128, mb), blk, 0, stream>>>(
      fusedb, wkb, kb, kbuf, nullptr, nullptr, nullptr, 0, M_, C_, KD_);
  gemm_wmma_bf16<EPI_F32, false><<<dim3(C_ / 128, mb), blk, 0, stream>>>(
      fusedb, wvb, vbp, vbuf, nullptr, nullptr, nullptr, 0, M_, C_, KD_);

  // ---- single-query cross attention + gate ----
  small_linear<0><<<(16 * 768 + 255) / 256, 256, 0, stream>>>(
      txtf, qw, qb, nullptr, qbuf, 16, 768, 768);
  attn_kernel<<<B_ * H_, 256, 0, stream>>>(qbuf, kbuf, vbuf, taskp, NTOK, C_, H_);
  small_linear<0><<<(16 * 768 + 255) / 256, 256, 0, stream>>>(
      taskp, aow, aob, nullptr, task, 16, 768, 768);
  small_linear<0><<<1, 256, 0, stream>>>(
      task, gw, gb, nullptr, gate_logits, 16, E_, 768);
  gate_softmax<<<1, 32, 0, stream>>>(gate_logits, gp, B_, E_);

  // ---- dense experts, gate-weighted accumulate into d_out ----
  for (int e = 0; e < E_; ++e) {
    gemm_wmma_bf16<EPI_QGELU_BF16, false><<<dim3(F_ / 128, mb), blk, 0, stream>>>(
        hidb, we1b + (size_t)e * F_ * D_, eb1 + (size_t)e * F_,
        nullptr, hbig, nullptr, nullptr, 0, M_, F_, D_);
    if (e == 0) {
      gemm_wmma_bf16<EPI_MOE, true><<<dim3(D_ / 128, mb), blk, 0, stream>>>(
          hbig, we2b + (size_t)e * D_ * F_, eb2 + (size_t)e * D_,
          out, nullptr, nullptr, gp, e, M_, D_, F_);
    } else {
      gemm_wmma_bf16<EPI_MOE, false><<<dim3(D_ / 128, mb), blk, 0, stream>>>(
          hbig, we2b + (size_t)e * D_ * F_, eb2 + (size_t)e * D_,
          out, nullptr, nullptr, gp, e, M_, D_, F_);
    }
  }
}
